// NDCN_67602785239432
// MI455X (gfx1250) — matrix operand order repaired and checked by reading, not compiled, not run on hardware
//
#include <hip/hip_runtime.h>
#include <hip/hip_bf16.h>

// ---------------------------------------------------------------------------
// NDCN forward on MI455X (gfx1250): bf16 WMMA GEMMs + f32 RK4 state.
// Shapes: N_NODES=4096, IN_DIM=512, HID=256, N_CLS=128, N_T=20.
//
// GEMM uses CDNA5 async-to-LDS copies (GLOBAL_LOAD_ASYNC_TO_LDS_B128,
// ASYNCcnt) with double-buffered LDS (K-step 64 -> 16 WMMAs per barrier) so
// the next K-tile streams in while the current one feeds
// v_wmma_f32_16x16x32_bf16.
// ---------------------------------------------------------------------------

typedef __bf16 bf16_t;
typedef __attribute__((ext_vector_type(16))) __bf16 v16bf;
typedef __attribute__((ext_vector_type(8)))  float  v8f;

#define NODES   4096
#define IN_DIM  512
#define HIDF    256
#define NCLS    128
#define NT      20
#define KB      64      // K-step per staged tile (2 x 32-deep WMMA sub-steps)

struct U4x2 { uint4 lo, hi; };

__device__ __forceinline__ v16bf make_frag(uint4 lo, uint4 hi) {
  U4x2 t{lo, hi};
  return __builtin_bit_cast(v16bf, t);
}

// Per-lane async copy of 16B global -> LDS (ASYNCcnt-tracked). The immediate
// offset is added to BOTH the LDS and the global address (ISA 15.18).
#define ASYNC_B128_OFF(offstr, ldsoff, gaddr)                                  \
  asm volatile("global_load_async_to_lds_b128 %0, %1, off offset:" offstr      \
               :: "v"(ldsoff), "v"((unsigned long long)(size_t)(gaddr))        \
               : "memory")

__device__ __forceinline__ void wait_async0() {
  asm volatile("s_wait_asynccnt 0x0" ::: "memory");
}

// generic shared-memory pointer -> LDS byte offset (low 32 bits of aperture addr)
__device__ __forceinline__ unsigned lds_off_of(const void* p) {
  return (unsigned)(size_t)p;
}

// ---------------------------------------------------------------------------
// f32 -> bf16 elementwise convert (grid-stride)
// ---------------------------------------------------------------------------
__global__ void ndcn_f32_to_bf16(const float* __restrict__ in,
                                 bf16_t* __restrict__ out, int n) {
  int i = blockIdx.x * blockDim.x + threadIdx.x;
  int stride = gridDim.x * blockDim.x;
  for (; i < n; i += stride) out[i] = (bf16_t)in[i];
}

// ---------------------------------------------------------------------------
// Generic GEMM: C = act(A[M,K](bf16 rm) * Bt[N,K]^T(bf16) + bias)
//   Tile: 64(M) x 128(N), 128 threads (4 waves), K-step 64 (2 WMMA sub-steps),
//   double-buffered LDS filled by async-to-LDS copies.
//   act: 0 = none, 1 = tanh, 2 = relu.  Outputs nullable.
// ---------------------------------------------------------------------------
__global__ __launch_bounds__(128)
void ndcn_gemm_bf16_wmma(const bf16_t* __restrict__ A,
                         const bf16_t* __restrict__ Bt,
                         const float*  __restrict__ bias,
                         float*  __restrict__ outF,
                         bf16_t* __restrict__ outBrm,
                         bf16_t* __restrict__ outBT, int ldT,
                         int N, int K, int act) {
  const int bm   = blockIdx.x * 64;
  const int bn   = blockIdx.y * 128;
  const int tid  = threadIdx.x;
  const int wave = tid >> 5;
  const int lane = tid & 31;
  const int l16  = lane & 15;
  const int kh   = lane >> 4;   // 0: lanes 0-15, 1: lanes 16-31

  __shared__ alignas(16) bf16_t sA[2][64 * KB];   // [buf][row][k]  2 x 8 KB
  __shared__ alignas(16) bf16_t sB[2][128 * KB];  // [buf][n][k]    2 x 16 KB

  // ---- per-thread copy mapping (async: one instr = 16B per lane)
  const int arow  = tid >> 1;          // A: 2 threads per row, 64B each
  const int ahalf = tid & 1;
  const bf16_t* gA0 = A + (size_t)(bm + arow) * K + ahalf * 32;
  const bf16_t* gB0 = Bt + (size_t)(bn + tid) * K;   // B: 128B per thread
  unsigned laA[2], laB[2];
  laA[0] = lds_off_of(&sA[0][arow * KB + ahalf * 32]);
  laA[1] = lds_off_of(&sA[1][arow * KB + ahalf * 32]);
  laB[0] = lds_off_of(&sB[0][tid * KB]);
  laB[1] = lds_off_of(&sB[1][tid * KB]);

  auto issue_tile = [&](int k0, int p) {
    const bf16_t* ga = gA0 + k0;
    ASYNC_B128_OFF("0",  laA[p], ga);
    ASYNC_B128_OFF("16", laA[p], ga);
    ASYNC_B128_OFF("32", laA[p], ga);
    ASYNC_B128_OFF("48", laA[p], ga);
    const bf16_t* gb = gB0 + k0;
    ASYNC_B128_OFF("0",   laB[p], gb);
    ASYNC_B128_OFF("16",  laB[p], gb);
    ASYNC_B128_OFF("32",  laB[p], gb);
    ASYNC_B128_OFF("48",  laB[p], gb);
    ASYNC_B128_OFF("64",  laB[p], gb);
    ASYNC_B128_OFF("80",  laB[p], gb);
    ASYNC_B128_OFF("96",  laB[p], gb);
    ASYNC_B128_OFF("112", laB[p], gb);
  };

  v8f acc[8];
  const v8f vzero = {0.f, 0.f, 0.f, 0.f, 0.f, 0.f, 0.f, 0.f};
#pragma unroll
  for (int c = 0; c < 8; ++c) acc[c] = vzero;

  // ---- prologue: stage tile 0 into buffer 0
  issue_tile(0, 0);

  int p = 0;
  for (int k0 = 0; k0 < K; k0 += KB) {
    wait_async0();        // own portion of tile p landed in LDS
    __syncthreads();      // all waves' portions landed; buf p^1 free to reuse

    if (k0 + KB < K) issue_tile(k0 + KB, p ^ 1);   // overlaps with compute

    const bf16_t* pa = &sA[p][0];
    const bf16_t* pb = &sB[p][0];
    const int ar = wave * 16 + l16;

#pragma unroll
    for (int kk = 0; kk < KB; kk += 32) {
      // ---- A fragment (16x32 bf16, contiguous k = 16*kh + s, same order as B)
      uint4 alo = *(const uint4*)(pa + ar * KB + kk + kh * 16);
      uint4 ahi = *(const uint4*)(pa + ar * KB + kk + kh * 16 + 8);
      v16bf afrag = make_frag(alo, ahi);

      // ---- 8 B fragments + WMMAs
#pragma unroll
      for (int c = 0; c < 8; ++c) {
        const uint4* bp = (const uint4*)(pb + (c * 16 + l16) * KB + kk + kh * 16);
        v16bf bfrag = make_frag(bp[0], bp[1]);
        acc[c] = __builtin_amdgcn_wmma_f32_16x16x32_bf16(
            false, afrag, false, bfrag, (short)0, acc[c], false, false);
      }
    }
    p ^= 1;
  }

  // ---- epilogue: D layout — VGPR j, lanes 0-15: M=j, lanes 16-31: M=j+8
#pragma unroll
  for (int c = 0; c < 8; ++c) {
    int ncol = bn + c * 16 + l16;
    float bv = bias ? bias[ncol] : 0.0f;
#pragma unroll
    for (int j = 0; j < 8; ++j) {
      float v = acc[c][j] + bv;
      if (act == 1)      v = tanhf(v);
      else if (act == 2) v = fmaxf(v, 0.0f);
      int r = bm + wave * 16 + kh * 8 + j;
      if (outF)   outF[(size_t)r * N + ncol] = v;
      if (outBrm) outBrm[(size_t)r * N + ncol] = (bf16_t)v;
      if (outBT)  outBT[(size_t)ncol * ldT + r] = (bf16_t)v;
    }
  }
}

// ---------------------------------------------------------------------------
// RK4 (3/8-rule) elementwise combines. dt read from vt on device (graph-safe).
//   mode 1: y = x + dt/3 * k1
//   mode 2: y = x + dt*(k2 - k1/3)
//   mode 3: y = x + dt*(k1 - k2 + k3)
//   mode 4: x' = x + dt*(k1 + 3k2 + 3k3 + k4)/8   (also writes f32 + bf16 rm)
// Always writes transposed bf16 y^T [HID][NODES] for the next A@y GEMM.
// ---------------------------------------------------------------------------
__global__ void ndcn_rk4_combine(const float* __restrict__ vt, int tstep, int mode,
                                 const float* __restrict__ x,
                                 const float* __restrict__ k1,
                                 const float* __restrict__ k2,
                                 const float* __restrict__ k3,
                                 const float* __restrict__ k4,
                                 float*  __restrict__ xout,
                                 bf16_t* __restrict__ xbrm,
                                 bf16_t* __restrict__ ybT) {
  int idx = blockIdx.x * blockDim.x + threadIdx.x;
  if (idx >= NODES * HIDF) return;
  float dt = vt[tstep + 1] - vt[tstep];
  float v = x[idx];
  if (mode == 1) {
    v += (dt * (1.0f / 3.0f)) * k1[idx];
  } else if (mode == 2) {
    v += dt * (k2[idx] - (1.0f / 3.0f) * k1[idx]);
  } else if (mode == 3) {
    v += dt * (k1[idx] - k2[idx] + k3[idx]);
  } else {
    v += (dt * 0.125f) * (k1[idx] + 3.0f * k2[idx] + 3.0f * k3[idx] + k4[idx]);
    xout[idx] = v;
    xbrm[idx] = (bf16_t)v;
  }
  int i = idx >> 8;          // node
  int j = idx & (HIDF - 1);  // feature
  ybT[(size_t)j * NODES + i] = (bf16_t)v;
}

// ---------------------------------------------------------------------------
// Host-side orchestration
// ---------------------------------------------------------------------------
extern "C" void kernel_launch(void* const* d_in, const int* in_sizes, int n_in,
                              void* d_out, int out_size, void* d_ws, size_t ws_size,
                              hipStream_t stream) {
  (void)in_sizes; (void)n_in; (void)out_size; (void)ws_size;

  const float* vt  = (const float*)d_in[0];
  const float* X   = (const float*)d_in[1];
  const float* Aad = (const float*)d_in[2];
  const float* W1  = (const float*)d_in[3];
  const float* b1  = (const float*)d_in[4];
  const float* W2  = (const float*)d_in[5];
  const float* b2  = (const float*)d_in[6];
  const float* Wt  = (const float*)d_in[7];
  const float* bt  = (const float*)d_in[8];
  const float* Wo  = (const float*)d_in[9];
  const float* bo  = (const float*)d_in[10];
  float* out = (float*)d_out;

  // ---- workspace carve-up (256B aligned)
  char* ws = (char*)d_ws;
  size_t off = 0;
  auto carve = [&](size_t bytes) -> void* {
    void* p = ws + off;
    off = (off + bytes + 255) & ~(size_t)255;
    return p;
  };
  bf16_t* Ab   = (bf16_t*)carve((size_t)NODES * NODES * 2);  // 32 MB, L2-resident
  bf16_t* Xb   = (bf16_t*)carve((size_t)NODES * IN_DIM * 2);
  bf16_t* W1b  = (bf16_t*)carve((size_t)HIDF * IN_DIM * 2);  // Bt layout == row-major W1
  bf16_t* W2b  = (bf16_t*)carve((size_t)HIDF * HIDF * 2);
  bf16_t* Wtb  = (bf16_t*)carve((size_t)HIDF * HIDF * 2);
  bf16_t* Wob  = (bf16_t*)carve((size_t)NCLS * HIDF * 2);
  bf16_t* T1b  = (bf16_t*)carve((size_t)NODES * HIDF * 2);   // tanh(X W1^T + b1)
  bf16_t* Gb   = (bf16_t*)carve((size_t)NODES * HIDF * 2);   // A @ y (bf16)
  bf16_t* xbrm = (bf16_t*)carve((size_t)NODES * HIDF * 2);   // state, bf16 row-major
  bf16_t* ybT  = (bf16_t*)carve((size_t)NODES * HIDF * 2);   // y^T bf16 [HID][NODES]
  float*  xs   = (float*)carve((size_t)NODES * HIDF * 4);    // state f32
  float*  k1   = (float*)carve((size_t)NODES * HIDF * 4);
  float*  k2   = (float*)carve((size_t)NODES * HIDF * 4);
  float*  k3   = (float*)carve((size_t)NODES * HIDF * 4);
  float*  k4   = (float*)carve((size_t)NODES * HIDF * 4);

  auto cvt = [&](const float* src, bf16_t* dst, size_t n) {
    int blocks = (int)((n + 1023) / 1024);
    if (blocks > 8192) blocks = 8192;
    ndcn_f32_to_bf16<<<dim3(blocks), dim3(1024), 0, stream>>>(src, dst, (int)n);
  };
  auto gemm = [&](const bf16_t* A_, const bf16_t* Bt_, const float* bias_,
                  float* oF, bf16_t* oBrm, bf16_t* oBT, int ldT_,
                  int M_, int N_, int K_, int act_) {
    dim3 grid(M_ / 64, N_ / 128);
    ndcn_gemm_bf16_wmma<<<grid, dim3(128), 0, stream>>>(
        A_, Bt_, bias_, oF, oBrm, oBT, ldT_, N_, K_, act_);
  };
  auto combine = [&](int t, int mode) {
    ndcn_rk4_combine<<<dim3((NODES * HIDF) / 256), dim3(256), 0, stream>>>(
        vt, t, mode, xs, k1, k2, k3, k4, xs, xbrm, ybT);
  };

  // ---- one-time (per call) bf16 conversions
  cvt(Aad, Ab,  (size_t)NODES * NODES);
  cvt(X,   Xb,  (size_t)NODES * IN_DIM);
  cvt(W1,  W1b, (size_t)HIDF * IN_DIM);
  cvt(W2,  W2b, (size_t)HIDF * HIDF);
  cvt(Wt,  Wtb, (size_t)HIDF * HIDF);
  cvt(Wo,  Wob, (size_t)NCLS * HIDF);

  // ---- input layer: h = tanh(X W1^T + b1) W2^T + b2
  gemm(Xb,  W1b, b1, nullptr, T1b, nullptr, 0, NODES, HIDF, IN_DIM, /*tanh*/1);
  gemm(T1b, W2b, b2, xs, xbrm, ybT, NODES, NODES, HIDF, HIDF, /*none*/0);

  // ---- out[0] = h Wo^T + bo
  gemm(xbrm, Wob, bo, out, nullptr, nullptr, 0, NODES, NCLS, HIDF, 0);

  // ---- RK4 (3/8 rule) over 19 intervals
  for (int t = 0; t < NT - 1; ++t) {
    // k1 = f(x)
    gemm(Ab, ybT, nullptr, nullptr, Gb, nullptr, 0, NODES, HIDF, NODES, 0);
    gemm(Gb, Wtb, bt, k1, nullptr, nullptr, 0, NODES, HIDF, HIDF, /*relu*/2);
    combine(t, 1);
    // k2 = f(x + dt*k1/3)
    gemm(Ab, ybT, nullptr, nullptr, Gb, nullptr, 0, NODES, HIDF, NODES, 0);
    gemm(Gb, Wtb, bt, k2, nullptr, nullptr, 0, NODES, HIDF, HIDF, 2);
    combine(t, 2);
    // k3 = f(x + dt*(k2 - k1/3))
    gemm(Ab, ybT, nullptr, nullptr, Gb, nullptr, 0, NODES, HIDF, NODES, 0);
    gemm(Gb, Wtb, bt, k3, nullptr, nullptr, 0, NODES, HIDF, HIDF, 2);
    combine(t, 3);
    // k4 = f(x + dt*(k1 - k2 + k3))
    gemm(Ab, ybT, nullptr, nullptr, Gb, nullptr, 0, NODES, HIDF, NODES, 0);
    gemm(Gb, Wtb, bt, k4, nullptr, nullptr, 0, NODES, HIDF, HIDF, 2);
    // x' = x + dt*(k1 + 3k2 + 3k3 + k4)/8 ; refresh xs / xbrm / ybT
    combine(t, 4);
    // out[t+1] = x' Wo^T + bo
    gemm(xbrm, Wob, bo, out + (size_t)(t + 1) * NODES * NCLS,
         nullptr, nullptr, 0, NODES, NCLS, HIDF, 0);
  }
}
